// TKANLayer_52639119180167
// MI455X (gfx1250) — matrix operand order repaired and checked by reading, not compiled, not run on hardware
//
#include <hip/hip_runtime.h>
#include <hip/hip_bf16.h>

typedef __attribute__((ext_vector_type(16))) __bf16 v16bf;
typedef __attribute__((ext_vector_type(8)))  float  v8f;

static constexpr int kB = 64, kT = 2048, kD = 128, kH = 256;
static constexpr float kEps = 1e-5f;

// ---- packed-weight offsets in bf16 elements inside workspace ----
// B-operand fragment layout (16-bit B 32x16, this code's contiguous k order): fragment f = nblk*nKblk+kblk,
// lane L holds 16 contiguous bf16: element e -> K = kblk*32 + (L>=16?16:0)+e, N = nblk*16 + (L&15)
static constexpr size_t OFF_W  = 0;                              // 4 x (128*256)
static constexpr size_t OFF_U  = OFF_W + 4 * (size_t)(kD * kH);  // 4 x (256*256)
static constexpr size_t OFF_K1 = OFF_U + 4 * (size_t)(kH * kH);  // 256*512
static constexpr size_t OFF_K2 = OFF_K1 + (size_t)(kH * 2 * kH); // 512*256

// -------- prep kernel: pack a (K x N) row-major f32 matrix into WMMA-B bf16 fragments --------
__global__ void tkan_pack_b(const float* __restrict__ src, __bf16* __restrict__ dst,
                            int K, int N) {
    int tid = blockIdx.x * blockDim.x + threadIdx.x;
    if (tid >= K * N) return;
    int e    = tid & 15;
    int lane = (tid >> 4) & 31;
    int f    = tid >> 9;                 // 512 elements per fragment
    int nKblk = K >> 5;
    int kblk  = f % nKblk;
    int nblk  = f / nKblk;
    int k = kblk * 32 + ((lane & 16) ? 16 : 0) + e;
    int n = nblk * 16 + (lane & 15);
    dst[tid] = (__bf16)src[(size_t)k * N + n];
}

// Inverse of the 16-bit A-matrix 16x32 layout (contiguous k order, matching B packing):
// value at (row r, k) lives at flat index (kblk*32 + lane)*16 + e with
//   lane = r + 16*bit4(k), e = k&15
__device__ inline int afrag_index(int r, int k) {
    int kblk = k >> 5, kk = k & 31;
    int lane = r + (kk & 16);
    int e    = kk & 15;
    return (kblk * 32 + lane) * 16 + e;
}

// A fragment = 16 contiguous bf16 per lane -> 2x ds_load_b128
__device__ inline v16bf load_a_frag(const __bf16* buf, int kblk, int lane) {
    return reinterpret_cast<const v16bf*>(buf)[kblk * 32 + lane];
}

__device__ inline v16bf load_b_frag(const __bf16* __restrict__ packed, int nKblk,
                                    int nblk, int kblk, int lane) {
    const v16bf* p = reinterpret_cast<const v16bf*>(packed);
    return p[(size_t)(nblk * nKblk + kblk) * 32 + lane];
}

__device__ inline float sigmoidf(float x) { return 1.0f / (1.0f + __expf(-x)); }

// -------- persistent scan kernel: 4 blocks x 512 threads (16 waves), block owns 16 batch rows --------
__launch_bounds__(512, 1)
__global__ void tkan_scan(const float* __restrict__ x,
                          const float* __restrict__ b_i, const float* __restrict__ b_f,
                          const float* __restrict__ b_c, const float* __restrict__ b_o,
                          const float* __restrict__ kb1, const float* __restrict__ kb2,
                          const float* __restrict__ gamma, const float* __restrict__ beta,
                          const __bf16* __restrict__ wpack,
                          float* __restrict__ out, float* __restrict__ h_out,
                          float* __restrict__ c_out) {
    // All activation staging is kept in WMMA A-fragment bf16 layout.
    __shared__ __align__(64) __bf16 s_xfrag[16 * kD];      // 4 KB
    __shared__ __align__(64) __bf16 s_hfrag[16 * kH];      // 8 KB
    __shared__ __align__(64) __bf16 s_cfrag[16 * kH];      // 8 KB
    __shared__ __align__(64) __bf16 s_tfrag[16 * 2 * kH];  // 16 KB
    __shared__ float s_mu[16], s_s2[16];

    const int tid    = threadIdx.x;
    const int lane   = tid & 31;
    const int w      = tid >> 5;          // wave id 0..15 = owned hidden-column tile
    const int b0     = blockIdx.x * 16;   // first batch row of this block
    const int nlocal = lane & 15;
    const int rbase  = (lane & 16) ? 8 : 0;
    const int ncol   = w * 16 + nlocal;   // owned hidden column (0..255)

    const __bf16* Wg[4] = { wpack + OFF_W + 0 * (size_t)(kD * kH),
                            wpack + OFF_W + 1 * (size_t)(kD * kH),
                            wpack + OFF_W + 2 * (size_t)(kD * kH),
                            wpack + OFF_W + 3 * (size_t)(kD * kH) };
    const __bf16* Ug[4] = { wpack + OFF_U + 0 * (size_t)(kH * kH),
                            wpack + OFF_U + 1 * (size_t)(kH * kH),
                            wpack + OFF_U + 2 * (size_t)(kH * kH),
                            wpack + OFF_U + 3 * (size_t)(kH * kH) };
    const __bf16* K1p = wpack + OFF_K1;
    const __bf16* K2p = wpack + OFF_K2;
    const float* biasg[4] = { b_i, b_f, b_c, b_o };

    // zero initial h, c fragments (bf16 0x0000 == 0.0)
    for (int i = tid; i < 16 * kH; i += blockDim.x) {
        s_hfrag[i] = (__bf16)0.0f;
        s_cfrag[i] = (__bf16)0.0f;
    }
    __syncthreads();

    const float gmv = gamma[ncol], btv = beta[ncol];
    // per-wave register state: c tile and latest h tile (rows rbase..rbase+7, col ncol)
    float c_reg[8], h_fin[8], o_t[8];
#pragma unroll
    for (int v = 0; v < 8; ++v) { c_reg[v] = 0.0f; h_fin[v] = 0.0f; }

    // precomputed A-frag scatter indices for this lane's owned tile (r = rbase+v, k = ncol)
    int aidx[8];
#pragma unroll
    for (int v = 0; v < 8; ++v) aidx[v] = afrag_index(rbase + v, ncol);

    for (int t = 0; t < kT; ++t) {
        // ---- stage x_t tile (16 x 128) straight into A-fragment layout; prefetch next ----
        for (int i = tid; i < 16 * kD; i += blockDim.x) {
            int r = i >> 7, d = i & (kD - 1);
            size_t gidx = ((size_t)(b0 + r) * kT + t) * kD + d;
            s_xfrag[afrag_index(r, d)] = (__bf16)x[gidx];
            if (t + 1 < kT) __builtin_prefetch(&x[gidx + kD], 0, 0);
        }
        __syncthreads();

        // ---- gate pre-activations: one A fragment live, 4 accumulators ----
        v8f acc[4] = { v8f{}, v8f{}, v8f{}, v8f{} };
#pragma unroll
        for (int kb = 0; kb < kD / 32; ++kb) {
            v16bf a = load_a_frag(s_xfrag, kb, lane);
#pragma unroll
            for (int g = 0; g < 4; ++g)
                acc[g] = __builtin_amdgcn_wmma_f32_16x16x32_bf16(
                    false, a, false, load_b_frag(Wg[g], kD / 32, w, kb, lane),
                    (short)0, acc[g], false, false);
        }
#pragma unroll
        for (int kb = 0; kb < kH / 32; ++kb) {
            v16bf a = load_a_frag(s_hfrag, kb, lane);
#pragma unroll
            for (int g = 0; g < 4; ++g)
                acc[g] = __builtin_amdgcn_wmma_f32_16x16x32_bf16(
                    false, a, false, load_b_frag(Ug[g], kH / 32, w, kb, lane),
                    (short)0, acc[g], false, false);
        }
        const float bi_s = biasg[0][ncol], bf_s = biasg[1][ncol];
        const float bc_s = biasg[2][ncol], bo_s = biasg[3][ncol];

        // ---- cell update in registers; publish c to A-frag LDS ----
#pragma unroll
        for (int v = 0; v < 8; ++v) {
            float ig = sigmoidf(acc[0][v] + bi_s);
            float fg = sigmoidf(acc[1][v] + bf_s);
            float cg = tanhf(acc[2][v] + bc_s);
            o_t[v]   = sigmoidf(acc[3][v] + bo_s);
            c_reg[v] = fg * c_reg[v] + ig * cg;
            s_cfrag[aidx[v]] = (__bf16)c_reg[v];
        }
        __syncthreads();

        // ---- KAN layer 1: t = tanh(c @ K1 + kb1), N=512; wave handles nblk 2w, 2w+1 ----
        {
            v8f a1[2] = { v8f{}, v8f{} };
#pragma unroll
            for (int kb = 0; kb < kH / 32; ++kb) {
                v16bf a = load_a_frag(s_cfrag, kb, lane);
#pragma unroll
                for (int j = 0; j < 2; ++j)
                    a1[j] = __builtin_amdgcn_wmma_f32_16x16x32_bf16(
                        false, a, false, load_b_frag(K1p, kH / 32, 2 * w + j, kb, lane),
                        (short)0, a1[j], false, false);
            }
#pragma unroll
            for (int j = 0; j < 2; ++j) {
                const int n1 = (2 * w + j) * 16 + nlocal;
                const float bb = kb1[n1];
#pragma unroll
                for (int v = 0; v < 8; ++v)
                    s_tfrag[afrag_index(rbase + v, n1)] = (__bf16)tanhf(a1[j][v] + bb);
            }
        }
        __syncthreads();

        // ---- KAN layer 2: ctr = t @ K2 + kb2, N=256; wave handles nblk w ----
        v8f acc2 = {};
#pragma unroll
        for (int kb = 0; kb < (2 * kH) / 32; ++kb) {
            v16bf a = load_a_frag(s_tfrag, kb, lane);
            acc2 = __builtin_amdgcn_wmma_f32_16x16x32_bf16(
                false, a, false, load_b_frag(K2p, (2 * kH) / 32, w, kb, lane),
                (short)0, acc2, false, false);
        }
        float hb[8];
        const float kb2v = kb2[ncol];
#pragma unroll
        for (int v = 0; v < 8; ++v)
            hb[v] = o_t[v] * tanhf(acc2[v] + kb2v);

        // ---- LayerNorm over H=256 per row ----
        if (tid < 16) { s_mu[tid] = 0.0f; s_s2[tid] = 0.0f; }
        __syncthreads();
#pragma unroll
        for (int v = 0; v < 8; ++v) {
            float s = hb[v], q = hb[v] * hb[v];
#pragma unroll
            for (int m = 8; m >= 1; m >>= 1) {   // reduce 16 cols within the half-wave
                s += __shfl_xor(s, m, 32);
                q += __shfl_xor(q, m, 32);
            }
            if (nlocal == 0) {
                atomicAdd(&s_mu[rbase + v], s);   // ds_add_f32
                atomicAdd(&s_s2[rbase + v], q);
            }
        }
        __syncthreads();
#pragma unroll
        for (int v = 0; v < 8; ++v) {
            const int r = rbase + v;
            float mu  = s_mu[r] * (1.0f / kH);
            float var = s_s2[r] * (1.0f / kH) - mu * mu;
            float hn  = (hb[v] - mu) * rsqrtf(var + kEps) * gmv + btv;
            h_fin[v] = hn;
            s_hfrag[aidx[v]] = (__bf16)hn;   // publish h for next step's A operand
            out[((size_t)(b0 + r) * kT + t) * kH + ncol] = hn;
        }
        __syncthreads();
    }

    // ---- final h, c straight from per-wave registers ----
#pragma unroll
    for (int v = 0; v < 8; ++v) {
        const int r = rbase + v;
        h_out[(size_t)(b0 + r) * kH + ncol] = h_fin[v];
        c_out[(size_t)(b0 + r) * kH + ncol] = c_reg[v];
    }
}

extern "C" void kernel_launch(void* const* d_in, const int* in_sizes, int n_in,
                              void* d_out, int out_size, void* d_ws, size_t ws_size,
                              hipStream_t stream) {
    const float* x   = (const float*)d_in[0];
    const float* Wi  = (const float*)d_in[1];  const float* bi = (const float*)d_in[2];
    const float* Wf  = (const float*)d_in[3];  const float* bf = (const float*)d_in[4];
    const float* Wc  = (const float*)d_in[5];  const float* bc = (const float*)d_in[6];
    const float* Wo  = (const float*)d_in[7];  const float* bo = (const float*)d_in[8];
    const float* Ui  = (const float*)d_in[9];
    const float* Uf  = (const float*)d_in[10];
    const float* Uc  = (const float*)d_in[11];
    const float* Uo  = (const float*)d_in[12];
    const float* K1  = (const float*)d_in[13]; const float* kb1 = (const float*)d_in[14];
    const float* K2  = (const float*)d_in[15]; const float* kb2 = (const float*)d_in[16];
    const float* gam = (const float*)d_in[17]; const float* bet = (const float*)d_in[18];

    __bf16* wpack = (__bf16*)d_ws;   // 1.25 MB of packed bf16 B-fragments

    auto pack = [&](const float* src, size_t off, int K, int N) {
        int total = K * N;
        tkan_pack_b<<<(total + 255) / 256, 256, 0, stream>>>(src, wpack + off, K, N);
    };
    pack(Wi, OFF_W + 0 * (size_t)(kD * kH), kD, kH);
    pack(Wf, OFF_W + 1 * (size_t)(kD * kH), kD, kH);
    pack(Wc, OFF_W + 2 * (size_t)(kD * kH), kD, kH);
    pack(Wo, OFF_W + 3 * (size_t)(kD * kH), kD, kH);
    pack(Ui, OFF_U + 0 * (size_t)(kH * kH), kH, kH);
    pack(Uf, OFF_U + 1 * (size_t)(kH * kH), kH, kH);
    pack(Uc, OFF_U + 2 * (size_t)(kH * kH), kH, kH);
    pack(Uo, OFF_U + 3 * (size_t)(kH * kH), kH, kH);
    pack(K1, OFF_K1, kH, 2 * kH);
    pack(K2, OFF_K2, 2 * kH, kH);

    float* out   = (float*)d_out;
    float* h_out = out + (size_t)kB * kT * kH;
    float* c_out = h_out + (size_t)kB * kH;

    tkan_scan<<<kB / 16, 512, 0, stream>>>(x, bi, bf, bc, bo, kb1, kb2, gam, bet,
                                           wpack, out, h_out, c_out);
}
